// TernaryLinear_89404039233678
// MI455X (gfx1250) — matrix operand order repaired and checked by reading, not compiled, not run on hardware
//
#include <hip/hip_runtime.h>

typedef _Float16 v4h  __attribute__((ext_vector_type(4)));
typedef _Float16 v8h  __attribute__((ext_vector_type(8)));
typedef _Float16 v16h __attribute__((ext_vector_type(16)));
typedef float    v8f  __attribute__((ext_vector_type(8)));

#if defined(__gfx1250__) &&                                          \
    __has_builtin(__builtin_amdgcn_global_load_async_to_lds_b128) && \
    __has_builtin(__builtin_amdgcn_s_wait_asynccnt)
#define USE_ASYNC_LDS 1
// Builtin signature (from hipcc diagnostic): args are pointers to 128-bit
// int vectors; src in global (AS1 / __device__), dst in LDS (AS3).
typedef int b128i __attribute__((vector_size(16)));
typedef __attribute__((address_space(1))) b128i* gptr_t;
typedef __attribute__((address_space(3))) b128i* lptr_t;
#else
#define USE_ASYNC_LDS 0
#endif

// ---------------------------------------------------------------------------
// Kernel 1: quantize latent fp32 weight to value set {v0,v1,v2} (ascending),
// matching jnp.argmin first-min tie-break (strict > at midpoints), emit f16.
// ---------------------------------------------------------------------------
__global__ void quantize_w_kernel(const float* __restrict__ w,
                                  const float* __restrict__ values,
                                  _Float16* __restrict__ wq,
                                  int n4) {
  int i = blockIdx.x * blockDim.x + threadIdx.x;
  if (i >= n4) return;
  const float v0 = values[0], v1 = values[1], v2 = values[2];
  const float t01 = 0.5f * (v0 + v1);
  const float t12 = 0.5f * (v1 + v2);
  float4 wv = ((const float4*)w)[i];
  float f0 = wv.x, f1 = wv.y, f2 = wv.z, f3 = wv.w;
  v4h o;
  o[0] = (_Float16)((f0 > t01) ? ((f0 > t12) ? v2 : v1) : v0);
  o[1] = (_Float16)((f1 > t01) ? ((f1 > t12) ? v2 : v1) : v0);
  o[2] = (_Float16)((f2 > t01) ? ((f2 > t12) ? v2 : v1) : v0);
  o[3] = (_Float16)((f3 > t01) ? ((f3 > t12) ? v2 : v1) : v0);
  ((v4h*)wq)[i] = o;
}

// ---------------------------------------------------------------------------
// Kernel 2: convert activations f32 -> f16 once (x is reused 16x over N tiles)
// ---------------------------------------------------------------------------
__global__ void cvt_x_kernel(const float* __restrict__ x,
                             _Float16* __restrict__ xh, int n4) {
  int i = blockIdx.x * blockDim.x + threadIdx.x;
  if (i >= n4) return;
  float4 v = ((const float4*)x)[i];
  v4h o;
  o[0] = (_Float16)v.x;
  o[1] = (_Float16)v.y;
  o[2] = (_Float16)v.z;
  o[3] = (_Float16)v.w;
  ((v4h*)xh)[i] = o;
}

// ---------------------------------------------------------------------------
// Kernel 3: C = A(f16,[M][K]) * B(f16,[N][K])^T + bias, f32 accumulate/out.
// Block tile 256x128, BK=32, 8 wave32s as 4Mx2N, 64x64 per wave
// (16 wmma : 16 ds_load_b128 per stage). Double-buffered LDS filled by
// GLOBAL_LOAD_ASYNC_TO_LDS_B128 (ASYNCcnt) when available.
// ---------------------------------------------------------------------------
#define BM 256
#define BN 128
#define BK 32
#define LDSW 40  // halves per row: BK + 8 pad -> 80B stride, 16B-aligned frags

__launch_bounds__(256)
__global__ void gemm_f16_wmma(const _Float16* __restrict__ A,
                              const _Float16* __restrict__ B,
                              const float* __restrict__ bias,
                              float* __restrict__ C,
                              int M, int N, int K) {
  __shared__ _Float16 sA[2][BM * LDSW];  // 2 * 20480 B
  __shared__ _Float16 sB[2][BN * LDSW];  // 2 * 10240 B   => 60 KB total

  const int tid  = threadIdx.x;
  const int lane = tid & 31;
  const int wave = tid >> 5;   // 0..7
  const int wm   = wave >> 1;  // 0..3 : which 64-row slice of the M tile
  const int wn   = wave & 1;   // 0..1 : which 64-col slice of the N tile

  const int  n0 = blockIdx.x * BN;
  const long m0 = (long)blockIdx.y * BM;

  const int lh = lane & 15;  // lane within half-wave
  const int kg = lane >> 4;  // k-group (0/1) per WMMA operand layout

  const v8f vzero = {};
  v8f acc[4][4];
#pragma unroll
  for (int mi = 0; mi < 4; ++mi)
#pragma unroll
    for (int ni = 0; ni < 4; ++ni) acc[mi][ni] = vzero;

  // Tile copy: A tile = 256 rows x 64 B = 1024 x16B chunks (4/thread),
  //            B tile = 128 rows x 64 B =  512 x16B chunks (2/thread).
  // chunk c: row = c>>2, half-offset = (c&3)*8.
#if USE_ASYNC_LDS
  auto stage = [&](int b, int kt) {
#pragma unroll
    for (int j = 0; j < 4; ++j) {
      const int c = tid + j * 256, row = c >> 2, p = (c & 3) * 8;
      __builtin_amdgcn_global_load_async_to_lds_b128(
          (gptr_t)(A + m0 * K + (long)kt * BK + (long)row * K + p),
          (lptr_t)(&sA[b][row * LDSW + p]), 0, 0);
    }
#pragma unroll
    for (int j = 0; j < 2; ++j) {
      const int c = tid + j * 256, row = c >> 2, p = (c & 3) * 8;
      __builtin_amdgcn_global_load_async_to_lds_b128(
          (gptr_t)(B + (long)n0 * K + (long)kt * BK + (long)row * K + p),
          (lptr_t)(&sB[b][row * LDSW + p]), 0, 0);
    }
  };
#else
  v8h rgA[4], rgB[2];
  auto gload = [&](int kt) {
    const _Float16* ag = A + m0 * K + (long)kt * BK;
    const _Float16* bg = B + (long)n0 * K + (long)kt * BK;
#pragma unroll
    for (int j = 0; j < 4; ++j) {
      const int c = tid + j * 256, row = c >> 2, p = (c & 3) * 8;
      rgA[j] = *(const v8h*)(ag + (long)row * K + p);
    }
#pragma unroll
    for (int j = 0; j < 2; ++j) {
      const int c = tid + j * 256, row = c >> 2, p = (c & 3) * 8;
      rgB[j] = *(const v8h*)(bg + (long)row * K + p);
    }
  };
  auto sstore = [&](int b) {
#pragma unroll
    for (int j = 0; j < 4; ++j) {
      const int c = tid + j * 256, row = c >> 2, p = (c & 3) * 8;
      *(v8h*)(&sA[b][row * LDSW + p]) = rgA[j];
    }
#pragma unroll
    for (int j = 0; j < 2; ++j) {
      const int c = tid + j * 256, row = c >> 2, p = (c & 3) * 8;
      *(v8h*)(&sB[b][row * LDSW + p]) = rgB[j];
    }
  };
#endif

  const int nsteps = K / BK;
#if USE_ASYNC_LDS
  stage(0, 0);
  __builtin_amdgcn_s_wait_asynccnt(0);
  __syncthreads();
#else
  gload(0);
  sstore(0);
  __syncthreads();
#endif

  int buf = 0;
  for (int kt = 0; kt < nsteps; ++kt) {
    const bool more = (kt + 1 < nsteps);
#if USE_ASYNC_LDS
    if (more) stage(buf ^ 1, kt + 1);  // DMA next stage, overlap with WMMA
#else
    if (more) gload(kt + 1);
#endif

    // A fragment (16-bit 16x32): lane l -> row l&15,
    //   halves {kb..kb+7, kb+16..kb+23}, kb = (l>>4)*8
    v16h af[4], bf[4];
#pragma unroll
    for (int mi = 0; mi < 4; ++mi) {
      const _Float16* p = &sA[buf][(wm * 64 + mi * 16 + lh) * LDSW + kg * 8];
      ((v8h*)&af[mi])[0] = *(const v8h*)(p);
      ((v8h*)&af[mi])[1] = *(const v8h*)(p + 16);
    }
    // B fragment (32x16): lane l -> column l&15, K {kb..kb+7, kb+16..kb+23}, kb = (l>>4)*8
    //   (B column n == row n of row-major wq, since we need x @ wq^T)
#pragma unroll
    for (int ni = 0; ni < 4; ++ni) {
      const _Float16* p = &sB[buf][(wn * 64 + ni * 16 + lh) * LDSW + kg * 8];
      ((v8h*)&bf[ni])[0] = *(const v8h*)(p);
      ((v8h*)&bf[ni])[1] = *(const v8h*)(p + 16);
    }

#pragma unroll
    for (int mi = 0; mi < 4; ++mi)
#pragma unroll
      for (int ni = 0; ni < 4; ++ni)
        acc[mi][ni] = __builtin_amdgcn_wmma_f32_16x16x32_f16(
            false, af[mi], false, bf[ni], (short)0, acc[mi][ni], false, false);

#if USE_ASYNC_LDS
    if (more) __builtin_amdgcn_s_wait_asynccnt(0);
    __syncthreads();
#else
    __syncthreads();
    if (more) {
      sstore(buf ^ 1);
      __syncthreads();
    }
#endif
    buf ^= 1;
  }

  // Epilogue. C/D layout: lane l, vgpr r -> n = l&15, m = r + (l>>4)*8
#pragma unroll
  for (int ni = 0; ni < 4; ++ni) {
    const int n = n0 + wn * 64 + ni * 16 + lh;
    const float bv = bias[n];
#pragma unroll
    for (int mi = 0; mi < 4; ++mi) {
      const long mb = m0 + wm * 64 + mi * 16 + (long)kg * 8;
      float* cp = C + mb * N + n;
#pragma unroll
      for (int r = 0; r < 8; ++r) cp[(long)r * N] = acc[mi][ni][r] + bv;
    }
  }
}

// ---------------------------------------------------------------------------
extern "C" void kernel_launch(void* const* d_in, const int* in_sizes, int n_in,
                              void* d_out, int out_size, void* d_ws, size_t ws_size,
                              hipStream_t stream) {
  const float* x      = (const float*)d_in[0];
  const float* w      = (const float*)d_in[1];
  const float* bias   = (const float*)d_in[2];
  const float* values = (const float*)d_in[3];

  const int  N = in_sizes[2];            // out_features (bias length) = 2048
  const int  K = in_sizes[1] / N;        // in_features = 2048
  const long M = (long)in_sizes[0] / K;  // tokens = 16384

  _Float16* wq = (_Float16*)d_ws;     // [N][K] f16 quantized weight
  _Float16* xh = wq + (size_t)N * K;  // [M][K] f16 activations

  {
    int n4 = (N * K) / 4;
    quantize_w_kernel<<<(n4 + 255) / 256, 256, 0, stream>>>(w, values, wq, n4);
  }
  {
    long n4 = (M * (long)K) / 4;
    cvt_x_kernel<<<(int)((n4 + 255) / 256), 256, 0, stream>>>(x, xh, (int)n4);
  }
  dim3 grid(N / BN, (int)(M / BM));
  gemm_f16_wmma<<<grid, 256, 0, stream>>>(xh, wq, bias, (float*)d_out,
                                          (int)M, N, K);
}
